// Q8LinearLora_61452392071883
// MI455X (gfx1250) — 16-bit and 8-bit matrix operand order repaired and checked by reading, compiled, not run on hardware
//
#include <hip/hip_runtime.h>
#include <stdint.h>

#define M_DIM 8192
#define K_DIM 4096
#define N_DIM 4096

#define BM 128
#define BN 128
#define BK 64
#define LDS_STRIDE 80
#define TILE_BYTES (128 * LDS_STRIDE)

typedef __attribute__((ext_vector_type(8)))  int    v8i;
typedef __attribute__((ext_vector_type(8)))  float  v8f;
typedef __attribute__((ext_vector_type(2)))  float  v2f;
typedef __attribute__((ext_vector_type(16))) __bf16 v16bf;
typedef __attribute__((ext_vector_type(4)))  unsigned int u32x4;
typedef __attribute__((ext_vector_type(8)))  int    i32x8;
typedef __attribute__((ext_vector_type(4)))  int    i32x4;

#if defined(__has_builtin)
#if __has_builtin(__builtin_amdgcn_tensor_load_to_lds) && \
    __has_builtin(__builtin_amdgcn_s_wait_tensorcnt)
#define USE_TDM 1
#endif
#endif
#ifndef USE_TDM
#define USE_TDM 0
#endif

static __device__ __forceinline__ float bf16_bits_to_f32(uint32_t b) {
  union { uint32_t u; float f; } c; c.u = b << 16; return c.f;
}
static __device__ __forceinline__ __bf16 f32_to_bf16(float f) {
  union { float f; uint32_t u; } c; c.f = f;
  uint32_t u = c.u + 0x7FFFu + ((c.u >> 16) & 1u);
  union { uint16_t s; __bf16 h; } o; o.s = (uint16_t)(u >> 16); return o.h;
}
static __device__ __forceinline__ __bf16 bits_to_bf16(uint16_t s) {
  union { uint16_t s; __bf16 h; } o; o.s = s; return o.h;
}
static __device__ __forceinline__ float gelu_tanh(float x) {
  float x3 = x * x * x;
  float t = tanhf(0.7978845608028654f * (x + 0.044715f * x3));
  return 0.5f * x * (1.0f + t);
}

#if USE_TDM
static __device__ __forceinline__ void tdm_load_tile(
    uint32_t lds_byte_off, const void* gptr,
    uint32_t tile_k, uint32_t tile_rows,
    uint32_t tensor_d0, uint32_t tensor_d1, uint32_t row_stride) {
  const uint64_t ga = (uint64_t)(uintptr_t)gptr;
  u32x4 g0;
  g0[0] = 1u;
  g0[1] = lds_byte_off;
  g0[2] = (uint32_t)ga;
  g0[3] = (uint32_t)((ga >> 32) & 0x01FFFFFFu) | (2u << 30);
  i32x8 g1;
  g1[0] = (int)((1u << 20) | (3u << 22) | (3u << 25));
  g1[1] = (int)((tensor_d0 & 0xFFFFu) << 16);
  g1[2] = (int)(((tensor_d0 >> 16) & 0xFFFFu) | ((tensor_d1 & 0xFFFFu) << 16));
  g1[3] = (int)(((tensor_d1 >> 16) & 0xFFFFu) | ((tile_k & 0xFFFFu) << 16));
  g1[4] = (int)(tile_rows & 0xFFFFu);
  g1[5] = (int)row_stride;
  g1[6] = 0;
  g1[7] = 0;
  i32x4 z;
  z[0] = 0; z[1] = 0; z[2] = 0; z[3] = 0;
  i32x8 z8;
  #pragma unroll
  for (int i = 0; i < 8; ++i) z8[i] = 0;
  __builtin_amdgcn_tensor_load_to_lds(g0, g1, z, z, z8, 0);
}
#endif

__global__ __launch_bounds__(32) void lora_proj_kernel(
    const int8_t* __restrict__ x, const float* __restrict__ xs,
    const uint16_t* __restrict__ lora_a, float* __restrict__ T) {
  const int lane = threadIdx.x & 31;
  const int hlf  = lane >> 4;
  const int l15  = lane & 15;
  const int m0   = blockIdx.x * 16;

  const int8_t*   xrow = x + (size_t)(m0 + l15) * K_DIM;
  const uint16_t* arow = lora_a + (size_t)l15 * K_DIM;
  const float     xscale = xs[m0 + l15];

  v8f acc;
  #pragma unroll
  for (int i = 0; i < 8; ++i) acc[i] = 0.0f;

  for (int kb = 0; kb < K_DIM; kb += 32) {
    v16bf a, b;
    const uint32_t* xp0 = (const uint32_t*)(xrow + kb + hlf * 16);
    const uint32_t* xp1 = (const uint32_t*)(xrow + kb + 8 + hlf * 16);
    uint32_t w0 = xp0[0], w1 = xp0[1], w2 = xp1[0], w3 = xp1[1];
    #pragma unroll
    for (int j = 0; j < 4; ++j) {
      a[j]      = f32_to_bf16((float)(int8_t)(w0 >> (8 * j)) * xscale);
      a[4 + j]  = f32_to_bf16((float)(int8_t)(w1 >> (8 * j)) * xscale);
      a[8 + j]  = f32_to_bf16((float)(int8_t)(w2 >> (8 * j)) * xscale);
      a[12 + j] = f32_to_bf16((float)(int8_t)(w3 >> (8 * j)) * xscale);
    }
    const uint32_t* bp = (const uint32_t*)(arow + kb + hlf * 16);
    #pragma unroll
    for (int j = 0; j < 8; ++j) {
      uint32_t w = bp[j];
      b[2 * j]     = bits_to_bf16((uint16_t)(w & 0xFFFFu));
      b[2 * j + 1] = bits_to_bf16((uint16_t)(w >> 16));
    }
    acc = __builtin_amdgcn_wmma_f32_16x16x32_bf16(false, a, false, b,
                                                  (short)0, acc, false, false);
  }
  #pragma unroll
  for (int v = 0; v < 8; ++v)
    T[(size_t)(m0 + hlf * 8 + v) * 16 + l15] = acc[v];
}

__global__ __launch_bounds__(256) void q8_gemm_kernel(
    const int8_t* __restrict__ x, const int8_t* __restrict__ w,
    const float* __restrict__ xs, const float* __restrict__ wscale,
    const float* __restrict__ bias, const uint16_t* __restrict__ lora_b,
    const float* __restrict__ T, float* __restrict__ out) {
  __shared__ unsigned char lds[2 * 2 * TILE_BYTES];

  const int tid  = threadIdx.x;
  const int lane = tid & 31;
  const int wid  = tid >> 5;
  const int wm   = wid >> 2;
  const int wn   = wid & 3;
  const int hlf  = lane >> 4;
  const int l15  = lane & 15;

  const int bm0 = blockIdx.y * BM;
  const int bn0 = blockIdx.x * BN;
  const int KT  = K_DIM / BK;

#if USE_TDM
  const uint32_t lds_base = (uint32_t)(uintptr_t)(&lds[0]);
  const int8_t* a_tile0 = x + (size_t)bm0 * K_DIM;
  const int8_t* b_tile0 = w + (size_t)bn0 * K_DIM;
  if (wid == 0) {
    tdm_load_tile(lds_base,              a_tile0, BK, BM, K_DIM, M_DIM, K_DIM);
    tdm_load_tile(lds_base + TILE_BYTES, b_tile0, BK, BN, K_DIM, N_DIM, K_DIM);
    __builtin_amdgcn_s_wait_tensorcnt(0);
  }
  __syncthreads();
#else
  const int srow = tid >> 1;
  const int soff = (tid & 1) * 32;
  const size_t a_base = (size_t)(bm0 + srow) * K_DIM + soff;
  const size_t b_base = (size_t)(bn0 + srow) * K_DIM + soff;
  const int lds_off = srow * LDS_STRIDE + soff;

  uint4 pa0, pa1, pb0, pb1;
  auto stage_load = [&](int k0) {
    const uint4* ap = (const uint4*)(x + a_base + k0);
    pa0 = ap[0]; pa1 = ap[1];
    const uint4* bp = (const uint4*)(w + b_base + k0);
    pb0 = bp[0]; pb1 = bp[1];
  };
  auto stage_store = [&](int buf) {
    uint4* as = (uint4*)(lds + buf * 2 * TILE_BYTES + lds_off);
    as[0] = pa0; as[1] = pa1;
    uint4* bs = (uint4*)(lds + buf * 2 * TILE_BYTES + TILE_BYTES + lds_off);
    bs[0] = pb0; bs[1] = pb1;
  };
  stage_load(0);
  stage_store(0);
  __syncthreads();
#endif

  v8i acc[4][2];
  #pragma unroll
  for (int mt = 0; mt < 4; ++mt)
    #pragma unroll
    for (int nt = 0; nt < 2; ++nt)
      #pragma unroll
      for (int i = 0; i < 8; ++i) acc[mt][nt][i] = 0;

  for (int kt = 0; kt < KT; ++kt) {
    const int cur = kt & 1;

#if USE_TDM
    if ((kt + 1 < KT) && (wid == 0)) {
      const int k0n = (kt + 1) * BK;
      const uint32_t nb = (uint32_t)((kt + 1) & 1) * 2u * TILE_BYTES;
      tdm_load_tile(lds_base + nb,              a_tile0 + k0n, BK, BM,
                    K_DIM, M_DIM, K_DIM);
      tdm_load_tile(lds_base + nb + TILE_BYTES, b_tile0 + k0n, BK, BN,
                    K_DIM, N_DIM, K_DIM);
    }
#else
    if (kt + 1 < KT) stage_load((kt + 1) * BK);
#endif

    const unsigned char* As = lds + cur * 2 * TILE_BYTES;
    const unsigned char* Bs = As + TILE_BYTES;

    v8i afrag[4], bfrag[2];
    #pragma unroll
    for (int mt = 0; mt < 4; ++mt) {
      const unsigned long long* p = (const unsigned long long*)
          (As + (wm * 64 + mt * 16 + l15) * LDS_STRIDE + hlf * 8);
      unsigned long long q0 = p[0], q1 = p[2], q2 = p[4], q3 = p[6];
      afrag[mt][0] = (int)(uint32_t)q0; afrag[mt][1] = (int)(q0 >> 32);
      afrag[mt][2] = (int)(uint32_t)q1; afrag[mt][3] = (int)(q1 >> 32);
      afrag[mt][4] = (int)(uint32_t)q2; afrag[mt][5] = (int)(q2 >> 32);
      afrag[mt][6] = (int)(uint32_t)q3; afrag[mt][7] = (int)(q3 >> 32);
    }
    #pragma unroll
    for (int nt = 0; nt < 2; ++nt) {
      const unsigned long long* p = (const unsigned long long*)
          (Bs + (wn * 32 + nt * 16 + l15) * LDS_STRIDE + hlf * 8);
      unsigned long long q0 = p[0], q1 = p[2], q2 = p[4], q3 = p[6];
      bfrag[nt][0] = (int)(uint32_t)q0; bfrag[nt][1] = (int)(q0 >> 32);
      bfrag[nt][2] = (int)(uint32_t)q1; bfrag[nt][3] = (int)(q1 >> 32);
      bfrag[nt][4] = (int)(uint32_t)q2; bfrag[nt][5] = (int)(q2 >> 32);
      bfrag[nt][6] = (int)(uint32_t)q3; bfrag[nt][7] = (int)(q3 >> 32);
    }
    #pragma unroll
    for (int mt = 0; mt < 4; ++mt)
      #pragma unroll
      for (int nt = 0; nt < 2; ++nt)
        acc[mt][nt] = __builtin_amdgcn_wmma_i32_16x16x64_iu8(
            true, afrag[mt], true, bfrag[nt], acc[mt][nt], false, false);

#if USE_TDM
    if ((kt + 1 < KT) && (wid == 0)) __builtin_amdgcn_s_wait_tensorcnt(0);
    __syncthreads();
#else
    if (kt + 1 < KT) stage_store((kt + 1) & 1);
    __syncthreads();
#endif
  }

  v2f ta[4][4];
  float xv[4][8];
  #pragma unroll
  for (int mt = 0; mt < 4; ++mt) {
    const int trow = bm0 + wm * 64 + mt * 16 + l15;
    #pragma unroll
    for (int s = 0; s < 4; ++s) {
      const float* tp = T + (size_t)trow * 16 + s * 4 + hlf * 2;
      ta[mt][s][0] = tp[0];
      ta[mt][s][1] = tp[1];
    }
    #pragma unroll
    for (int v = 0; v < 8; ++v)
      xv[mt][v] = xs[bm0 + wm * 64 + mt * 16 + hlf * 8 + v];
  }

  #pragma unroll
  for (int nt = 0; nt < 2; ++nt) {
    const int n = bn0 + wn * 32 + nt * 16 + l15;
    const float wsn = wscale[n];
    const float bnv = bias[n];
    v2f bb[4];
    #pragma unroll
    for (int s = 0; s < 4; ++s) {
      uint32_t wbits = *(const uint32_t*)(lora_b + (size_t)n * 16 + s * 4 + hlf * 2);
      bb[s][0] = bf16_bits_to_f32(wbits & 0xFFFFu);
      bb[s][1] = bf16_bits_to_f32(wbits >> 16);
    }
    #pragma unroll
    for (int mt = 0; mt < 4; ++mt) {
      v8f lf;
      #pragma unroll
      for (int i = 0; i < 8; ++i) lf[i] = 0.0f;
      #pragma unroll
      for (int s = 0; s < 4; ++s)
        lf = __builtin_amdgcn_wmma_f32_16x16x4_f32(false, ta[mt][s], false, bb[s],
                                                   (short)0, lf, false, false);
      #pragma unroll
      for (int v = 0; v < 8; ++v) {
        const int m = bm0 + wm * 64 + mt * 16 + hlf * 8 + v;
        float val = (float)acc[mt][nt][v] * xv[mt][v] * wsn + lf[v] + bnv;
        out[(size_t)m * N_DIM + n] = gelu_tanh(val);
      }
    }
  }
}

extern "C" void kernel_launch(void* const* d_in, const int* in_sizes, int n_in,
                              void* d_out, int out_size, void* d_ws, size_t ws_size,
                              hipStream_t stream) {
  (void)in_sizes; (void)n_in; (void)out_size; (void)ws_size;
  const int8_t*   x      = (const int8_t*)d_in[0];
  const int8_t*   w      = (const int8_t*)d_in[1];
  const float*    xs     = (const float*)d_in[2];
  const float*    wscale = (const float*)d_in[3];
  const float*    bias   = (const float*)d_in[4];
  const uint16_t* lora_a = (const uint16_t*)d_in[5];
  const uint16_t* lora_b = (const uint16_t*)d_in[6];
  float* out = (float*)d_out;
  float* T   = (float*)d_ws;

  lora_proj_kernel<<<M_DIM / 16, 32, 0, stream>>>(x, xs, lora_a, T);

  dim3 grid(N_DIM / BN, M_DIM / BM);
  q8_gemm_kernel<<<grid, 256, 0, stream>>>(x, w, xs, wscale, bias, lora_b, T, out);
}
